// GatherDiagonal2_40656160424521
// MI455X (gfx1250) — matrix operand order repaired and checked by reading, not compiled, not run on hardware
//
#include <hip/hip_runtime.h>

// GatherDiagonal2: per-diagonal channel mixing == 81 GEMMs (M=4096,N=256,K=256).
// bf16 hi/lo split WMMA (fp32-class accuracy, ~2.7x cheaper than f32 WMMA path),
// double-buffered LDS software pipeline: prefetch(global->VGPR) overlaps WMMA,
// one barrier per k-step.

typedef __attribute__((ext_vector_type(16))) __bf16 v16bf;
typedef __attribute__((ext_vector_type(8)))  __bf16 v8bf;
typedef __attribute__((ext_vector_type(8)))  float  v8f;

#define BATCH  4096
#define CIN    256
#define COUT   256
#define CELLS  81
#define KS     32      // K per step (bf16 WMMA K)
#define MT     128     // block tile M (batch)
#define NT     128     // block tile N (c_out)
#define PITCH  40      // bf16 elements per LDS row (80B; bank-spread, keeps 16B align)
#define NSTEPS (CIN / KS)

__global__ __launch_bounds__(256)
void diag_linear_kernel(const float* __restrict__ inp,
                        const float* __restrict__ wgt,
                        const float* __restrict__ bias,
                        float* __restrict__ out) {
  __shared__ __bf16 lAhi[2][MT * PITCH];   // 8 planes x 10KB = 80KB (<320KB/WGP)
  __shared__ __bf16 lAlo[2][MT * PITCH];
  __shared__ __bf16 lBhi[2][NT * PITCH];
  __shared__ __bf16 lBlo[2][NT * PITCH];

  const int cell = blockIdx.x;                 // cell-fastest: L2 reuse of input lines
  const int d    = (cell / 9) + (cell % 9);    // diagonal index 0..16
  const int m0   = blockIdx.y * MT;
  const int n0   = blockIdx.z * NT;

  const int t     = threadIdx.x;
  const int lane  = t & 31;
  const int wave  = t >> 5;
  const int waveM = wave & 3;                  // 4 waves along M (32 rows each)
  const int waveN = wave >> 2;                 // 2 waves along N (64 cols each)

  const int rstg  = t >> 1;                    // staging row (A: batch, B: c_out)
  const int cbase = (t & 1) * 16;              // staging K base within a KS slab

  const float* aBase = inp + (size_t)(m0 + rstg) * (CIN * CELLS)
                           + (size_t)cbase * CELLS + cell;
  const float* wBase = wgt + (size_t)d * COUT * CIN
                           + (size_t)(n0 + rstg) * CIN + cbase;

  float ra[16], rb[16];

  // ---- prologue: fetch tile 0 and publish into buffer 0 ----
  #pragma unroll
  for (int e = 0; e < 16; ++e) ra[e] = aBase[e * CELLS];
  {
    const float4* wp = (const float4*)wBase;
    #pragma unroll
    for (int q = 0; q < 4; ++q) {
      float4 v = wp[q];
      rb[q * 4 + 0] = v.x; rb[q * 4 + 1] = v.y;
      rb[q * 4 + 2] = v.z; rb[q * 4 + 3] = v.w;
    }
  }
  #pragma unroll
  for (int e = 0; e < 16; ++e) {
    __bf16 h = (__bf16)ra[e];
    lAhi[0][rstg * PITCH + cbase + e] = h;
    lAlo[0][rstg * PITCH + cbase + e] = (__bf16)(ra[e] - (float)h);
    __bf16 g = (__bf16)rb[e];
    lBhi[0][rstg * PITCH + cbase + e] = g;
    lBlo[0][rstg * PITCH + cbase + e] = (__bf16)(rb[e] - (float)g);
  }
  __syncthreads();

  v8f acc[2][4];
  #pragma unroll
  for (int i = 0; i < 2; ++i)
    #pragma unroll
    for (int j = 0; j < 4; ++j)
      #pragma unroll
      for (int r = 0; r < 8; ++r) acc[i][j][r] = 0.0f;

  const int aoff = (lane >> 4) * 16;  // A frag: lanes 0-15 K0..15,      16-31 K16..31
  const int boff = (lane >> 4) * 16;  // B frag: lanes 0-15 K0..15,      16-31 K16..31

  for (int ks = 0; ks < NSTEPS; ++ks) {
    const int p = ks & 1;
    const bool more = (ks + 1) < NSTEPS;

    // ---- prefetch tile ks+1 (global -> VGPR); in flight under the WMMAs ----
    if (more) {
      const float* ap = aBase + (size_t)(ks + 1) * KS * CELLS;
      #pragma unroll
      for (int e = 0; e < 16; ++e) ra[e] = ap[e * CELLS];
      const float4* wp = (const float4*)(wBase + (ks + 1) * KS);
      #pragma unroll
      for (int q = 0; q < 4; ++q) {
        float4 v = wp[q];
        rb[q * 4 + 0] = v.x; rb[q * 4 + 1] = v.y;
        rb[q * 4 + 2] = v.z; rb[q * 4 + 3] = v.w;
      }
    }

    // ---- fragments from LDS buffer p (16B ds_load_b128-shaped reads) ----
    v16bf Ahi[2], Alo[2], Bhi[4], Blo[4];
    #pragma unroll
    for (int sm = 0; sm < 2; ++sm) {
      const int mrow = waveM * 32 + sm * 16 + (lane & 15);
      const __bf16* pa = &lAhi[p][mrow * PITCH + aoff];
      v8bf a0 = *(const v8bf*)pa;
      v8bf a1 = *(const v8bf*)(pa + 8);
      Ahi[sm] = __builtin_shufflevector(a0, a1, 0,1,2,3,4,5,6,7,8,9,10,11,12,13,14,15);
      const __bf16* pl = &lAlo[p][mrow * PITCH + aoff];
      v8bf b0 = *(const v8bf*)pl;
      v8bf b1 = *(const v8bf*)(pl + 8);
      Alo[sm] = __builtin_shufflevector(b0, b1, 0,1,2,3,4,5,6,7,8,9,10,11,12,13,14,15);
    }
    #pragma unroll
    for (int sn = 0; sn < 4; ++sn) {
      const int ncol = waveN * 64 + sn * 16 + (lane & 15);
      const __bf16* pb = &lBhi[p][ncol * PITCH + boff];
      v8bf a0 = *(const v8bf*)pb;
      v8bf a1 = *(const v8bf*)(pb + 8);
      Bhi[sn] = __builtin_shufflevector(a0, a1, 0,1,2,3,4,5,6,7,8,9,10,11,12,13,14,15);
      const __bf16* pl = &lBlo[p][ncol * PITCH + boff];
      v8bf b0 = *(const v8bf*)pl;
      v8bf b1 = *(const v8bf*)(pl + 8);
      Blo[sn] = __builtin_shufflevector(b0, b1, 0,1,2,3,4,5,6,7,8,9,10,11,12,13,14,15);
    }

    // ---- bf16x3 split-precision MMA: hi*hi + hi*lo + lo*hi ----
    #pragma unroll
    for (int sm = 0; sm < 2; ++sm)
      #pragma unroll
      for (int sn = 0; sn < 4; ++sn) {
        acc[sm][sn] = __builtin_amdgcn_wmma_f32_16x16x32_bf16(
            false, Ahi[sm], false, Bhi[sn], (short)0, acc[sm][sn], false, false);
        acc[sm][sn] = __builtin_amdgcn_wmma_f32_16x16x32_bf16(
            false, Ahi[sm], false, Blo[sn], (short)0, acc[sm][sn], false, false);
        acc[sm][sn] = __builtin_amdgcn_wmma_f32_16x16x32_bf16(
            false, Alo[sm], false, Bhi[sn], (short)0, acc[sm][sn], false, false);
      }

    // ---- publish tile ks+1 into buffer p^1 (readers are on buffer p) ----
    if (more) {
      const int q = p ^ 1;
      #pragma unroll
      for (int e = 0; e < 16; ++e) {
        __bf16 h = (__bf16)ra[e];
        lAhi[q][rstg * PITCH + cbase + e] = h;
        lAlo[q][rstg * PITCH + cbase + e] = (__bf16)(ra[e] - (float)h);
        __bf16 g = (__bf16)rb[e];
        lBhi[q][rstg * PITCH + cbase + e] = g;
        lBlo[q][rstg * PITCH + cbase + e] = (__bf16)(rb[e] - (float)g);
      }
    }
    __syncthreads();   // single barrier per k-step: publishes p^1, retires p
  }

  // ---- epilogue: fused bias add, scatter store (merged in L2 across cells) ----
  const float* bptr = bias + (size_t)d * COUT;
  #pragma unroll
  for (int sn = 0; sn < 4; ++sn) {
    const int ncol = n0 + waveN * 64 + sn * 16 + (lane & 15);
    const float bv = bptr[ncol];
    const int mbase = m0 + waveM * 32 + (lane >> 4) * 8;  // C/D layout: lanes>=16 -> M+8
    #pragma unroll
    for (int sm = 0; sm < 2; ++sm)
      #pragma unroll
      for (int r = 0; r < 8; ++r) {
        out[(size_t)(mbase + sm * 16 + r) * (COUT * CELLS)
            + (size_t)ncol * CELLS + cell] = acc[sm][sn][r] + bv;
      }
  }
}

extern "C" void kernel_launch(void* const* d_in, const int* in_sizes, int n_in,
                              void* d_out, int out_size, void* d_ws, size_t ws_size,
                              hipStream_t stream) {
  const float* inp  = (const float*)d_in[0];   // [4096, 256, 9, 9]
  const float* wgt  = (const float*)d_in[1];   // [17, 256, 256]
  const float* bias = (const float*)d_in[2];   // [17, 256]
  float* out = (float*)d_out;                  // [4096, 256, 9, 9]

  dim3 grid(CELLS, BATCH / MT, COUT / NT);     // cell fastest -> L2 line reuse
  dim3 block(256);
  diag_linear_kernel<<<grid, block, 0, stream>>>(inp, wgt, bias, out);
}
